// TLSTMWithID_15247133901430
// MI455X (gfx1250) — compile-verified, matrix operand order checked by reading, not run on hardware
//
#include <hip/hip_runtime.h>

// ---------------- problem constants ----------------
#define Bsz 512
#define Ssz 512
#define Fsz 128
#define Hsz 256
#define Esz 16
#define FIN 143     // F-1+E
#define FINP 160    // padded to multiple of 32 for WMMA K
#define G4H 1024    // 4*H
#define KREC 512    // H (emb part) + H (h part)

typedef __bf16 bf16_t;
typedef __attribute__((ext_vector_type(16))) __bf16 v16bf;
typedef __attribute__((ext_vector_type(8)))  float  v8f;
typedef __attribute__((address_space(3))) void lds_void;

struct bfrag { uint4 a, b; };   // 32B = 16 bf16, 16B-aligned

__device__ __forceinline__ v16bf load_frag(const bf16_t* p) {
    bfrag f = *(const bfrag*)p;
    return __builtin_bit_cast(v16bf, f);
}

__device__ __forceinline__ v8f wmma_bf16(v16bf a, v16bf b, v8f c) {
    return __builtin_amdgcn_wmma_f32_16x16x32_bf16(
        /*neg_a=*/false, a, /*neg_b=*/false, b,
        /*c_mod=*/(short)0, c, /*reuse_a=*/false, /*reuse_b=*/false);
}

__device__ __forceinline__ float sigmoidf_(float x) {
    return 1.0f / (1.0f + __expf(-x));
}

// Async global->LDS copy of 32 contiguous bytes per lane (2 x B128).
// Both INST_OFFSETs apply to LDS dest and global src simultaneously
// per CDNA5 ISA 10.x/15.18.
__device__ __forceinline__ void async_copy_32B(const bf16_t* g, bf16_t* l) {
    asm volatile(
        "global_load_async_to_lds_b128 %0, %1, off\n\t"
        "global_load_async_to_lds_b128 %0, %1, off offset:16"
        :
        : "v"((lds_void*)l), "v"(g)
        : "memory");
}
__device__ __forceinline__ void wait_async_le2() {
    asm volatile("s_wait_asynccnt 0x2" ::: "memory");
}

// ---------------- kernel 0: pack weights to bf16 ----------------
__global__ void convert_kernel(const float* __restrict__ W_in,
                               const float* __restrict__ W_ih,
                               const float* __restrict__ W_hh,
                               const float* __restrict__ b_ih,
                               const float* __restrict__ b_hh,
                               bf16_t* __restrict__ Winp,
                               bf16_t* __restrict__ Wcat,
                               float*  __restrict__ bcat)
{
    int i = blockIdx.x * blockDim.x + threadIdx.x;
    int nthr = gridDim.x * blockDim.x;
    for (int idx = i; idx < Hsz * FINP; idx += nthr) {
        int n = idx / FINP, k = idx % FINP;
        Winp[idx] = (bf16_t)((k < FIN) ? W_in[(size_t)n * FIN + k] : 0.0f);
    }
    for (int idx = i; idx < G4H * KREC; idx += nthr) {
        int n = idx / KREC, k = idx % KREC;
        float v = (k < Hsz) ? W_ih[(size_t)n * Hsz + k]
                            : W_hh[(size_t)n * Hsz + (k - Hsz)];
        Wcat[idx] = (bf16_t)v;
    }
    for (int idx = i; idx < G4H; idx += nthr)
        bcat[idx] = b_ih[idx] + b_hh[idx];
}

// ---------------- kernel 1: embedded = tanh(fcat @ W_in^T + b_in) -> bf16 --
__global__ void __launch_bounds__(128)
embed_kernel(const float* __restrict__ x,
             const int*   __restrict__ id_indices,
             const float* __restrict__ id_emb,
             const float* __restrict__ b_in,
             const bf16_t* __restrict__ Winp,
             bf16_t* __restrict__ emb)
{
    __shared__ bf16_t A[16][168];
    const int tile = blockIdx.x;
    const int r0   = tile * 16;          // row in flattened (B*S)
    const int bidx = r0 / Ssz;           // whole tile shares one batch index
    const int tid  = threadIdx.x;
    const int id   = id_indices[bidx];

    for (int idx = tid; idx < 16 * 168; idx += 128) {
        int r = idx / 168, c = idx % 168;
        float v = 0.0f;
        if (c < Fsz - 1)            v = x[((size_t)(r0 + r)) * Fsz + c];
        else if (c < Fsz - 1 + Esz) v = id_emb[(size_t)id * Esz + (c - (Fsz - 1))];
        A[r][c] = (bf16_t)v;
    }
    __syncthreads();

    const int lane = tid & 31;
    const int wave = tid >> 5;
    const int lrow = lane & 15;
    const int lhi  = lane >> 4;

    v8f acc[4] = {};
    #pragma unroll
    for (int k0 = 0; k0 < FINP; k0 += 32) {
        v16bf a = load_frag(&A[lrow][k0 + 16 * lhi]);
        #pragma unroll
        for (int j = 0; j < 4; ++j) {
            int n = (wave * 4 + j) * 16 + lrow;
            v16bf bm = load_frag(&Winp[(size_t)n * FINP + k0 + 16 * lhi]);
            acc[j] = wmma_bf16(a, bm, acc[j]);
        }
    }
    #pragma unroll
    for (int j = 0; j < 4; ++j) {
        int col = (wave * 4 + j) * 16 + lrow;
        float bias = b_in[col];
        #pragma unroll
        for (int v = 0; v < 8; ++v) {
            int m = v + 8 * lhi;
            float val = tanhf(acc[j][v] + bias);
            emb[((size_t)(r0 + m)) * Hsz + col] = (bf16_t)val;
        }
    }
}

// ---------------- kernel 2: persistent recurrent scan --------------------
// 32 blocks (16-row batch tile each), 256 threads = 8 waves.
// Wave w owns h-column tiles {2w,2w+1} + their i/f/g/o gate tiles; cell state
// stays in D-fragment VGPRs; embedded_t tiles double-buffered in LDS via
// async global->LDS loads issued one step ahead.
__global__ void __launch_bounds__(256)
recurrent_kernel(const float* __restrict__ x,
                 const int*   __restrict__ lengths,
                 const bf16_t* __restrict__ emb,
                 const bf16_t* __restrict__ Wcat,
                 const float* __restrict__ bcat,
                 const float* __restrict__ W_dec,
                 const float* __restrict__ b_dec,
                 const float* __restrict__ W_out,
                 const float* __restrict__ b_out,
                 float* __restrict__ out)
{
    __shared__ bf16_t Ebuf[2][16][264];  // double-buffered embedded_t tile
    __shared__ bf16_t Hbuf[16][264];     // h_{t} (bf16) staging
    __shared__ float  hfin[16][264];     // final h for output head
    __shared__ float  dtv[16];

    const int b0   = blockIdx.x * 16;
    const int tid  = threadIdx.x;
    const int lane = tid & 31;
    const int wave = tid >> 5;
    const int lrow = lane & 15;
    const int lhi  = lane >> 4;
    const int col0 = (wave * 2 + 0) * 16 + lrow;
    const int col1 = (wave * 2 + 1) * 16 + lrow;

    const int rowA = tid >> 4;           // staging row (0..15)
    const int colA = (tid & 15) * 16;    // staging col base (0..240)

    // kick off async load of embedded tile for t=0 into Ebuf[0]
    async_copy_32B(&emb[(((size_t)(b0 + rowA)) * Ssz + 0) * Hsz + colA],
                   &Ebuf[0][rowA][colA]);

    // h0 = 0
    for (int idx = tid; idx < 16 * 256; idx += 256)
        Hbuf[idx / 256][idx % 256] = (bf16_t)0.0f;

    // per-lane loop-invariant scalars
    const float wd0 = W_dec[col0], wd1 = W_dec[col1];
    const float bd0 = b_dec[col0], bd1 = b_dec[col1];
    float bg[8];
    #pragma unroll
    for (int g = 0; g < 4; ++g) {
        bg[2 * g + 0] = bcat[g * Hsz + col0];
        bg[2 * g + 1] = bcat[g * Hsz + col1];
    }
    int leni[8];
    #pragma unroll
    for (int v = 0; v < 8; ++v) leni[v] = lengths[b0 + v + 8 * lhi];

    v8f c0v = {}, c1v = {};              // cell-state fragments

    for (int t = 0; t < Ssz; ++t) {
        const int par = t & 1;
        // prefetch embedded tile for t+1 into the other buffer (safe: all
        // waves passed the t-1 end barrier, so its reads are complete)
        const int tnext = (t + 1 < Ssz) ? (t + 1) : (Ssz - 1);
        async_copy_32B(&emb[(((size_t)(b0 + rowA)) * Ssz + tnext) * Hsz + colA],
                       &Ebuf[par ^ 1][rowA][colA]);
        // wait for step-t tile (the 2 oldest of <=4 outstanding async ops)
        wait_async_le2();
        if (tid < 16)
            dtv[tid] = x[(((size_t)(b0 + tid)) * Ssz + t) * Fsz + (Fsz - 1)];
        __syncthreads();                 // Ebuf[par], Hbuf, dtv all ready

        float dtr[8];
        #pragma unroll
        for (int v = 0; v < 8; ++v) dtr[v] = dtv[v + 8 * lhi];

        // gates = [emb_t | h] (16x512) @ Wcat^T for this wave's 8 n-tiles
        v8f acc[8] = {};                 // [gate(4)][tile(2)]
        #pragma unroll
        for (int k0 = 0; k0 < KREC; k0 += 32) {
            const bf16_t* ap = (k0 < Hsz)
                ? &Ebuf[par][lrow][k0 + 16 * lhi]
                : &Hbuf[lrow][(k0 - Hsz) + 16 * lhi];
            v16bf a = load_frag(ap);
            #pragma unroll
            for (int g = 0; g < 4; ++g) {
                #pragma unroll
                for (int jj = 0; jj < 2; ++jj) {
                    int n = (g * 16 + wave * 2 + jj) * 16 + lrow;
                    v16bf bm = load_frag(&Wcat[(size_t)n * KREC + k0 + 16 * lhi]);
                    acc[2 * g + jj] = wmma_bf16(a, bm, acc[2 * g + jj]);
                }
            }
        }
        __syncthreads();                 // all Ebuf/Hbuf reads complete

        // LSTM elementwise update in fragment layout
        #pragma unroll
        for (int v = 0; v < 8; ++v) {
            int   m   = v + 8 * lhi;
            float dt  = dtr[v];
            float msk = (t < leni[v]) ? 1.0f : 0.0f;
            {   // tile 0
                float gamma = sigmoidf_(dt * wd0 + bd0);
                float c  = c0v[v] * gamma;
                float gi = sigmoidf_(acc[0][v] + bg[0]);
                float gf = sigmoidf_(acc[2][v] + bg[2]);
                float gg = tanhf    (acc[4][v] + bg[4]);
                float go = sigmoidf_(acc[6][v] + bg[6]);
                c = gf * c + gi * gg;
                float h = go * tanhf(c);
                h *= msk; c *= msk;
                c0v[v] = c;
                Hbuf[m][col0] = (bf16_t)h;
                if (t == Ssz - 1) hfin[m][col0] = h;
            }
            {   // tile 1
                float gamma = sigmoidf_(dt * wd1 + bd1);
                float c  = c1v[v] * gamma;
                float gi = sigmoidf_(acc[1][v] + bg[1]);
                float gf = sigmoidf_(acc[3][v] + bg[3]);
                float gg = tanhf    (acc[5][v] + bg[5]);
                float go = sigmoidf_(acc[7][v] + bg[7]);
                c = gf * c + gi * gg;
                float h = go * tanhf(c);
                h *= msk; c *= msk;
                c1v[v] = c;
                Hbuf[m][col1] = (bf16_t)h;
                if (t == Ssz - 1) hfin[m][col1] = h;
            }
        }
    }
    __syncthreads();

    // output head: out = h_T @ W_out^T + b_out  (16 rows x 3 classes)
    if (tid < 48) {
        int r = tid / 3, cc = tid % 3;
        float s = b_out[cc];
        for (int k = 0; k < Hsz; ++k)
            s += hfin[r][k] * W_out[(size_t)cc * Hsz + k];
        out[(size_t)(b0 + r) * 3 + cc] = s;
    }
}

// ---------------- launcher ----------------
extern "C" void kernel_launch(void* const* d_in, const int* in_sizes, int n_in,
                              void* d_out, int out_size, void* d_ws, size_t ws_size,
                              hipStream_t stream) {
    const float* x          = (const float*)d_in[0];
    const int*   lengths    = (const int*)  d_in[1];
    const int*   id_indices = (const int*)  d_in[2];
    const float* id_emb     = (const float*)d_in[3];
    const float* W_in       = (const float*)d_in[4];
    const float* b_in       = (const float*)d_in[5];
    const float* W_dec      = (const float*)d_in[6];
    const float* b_dec      = (const float*)d_in[7];
    const float* W_ih       = (const float*)d_in[8];
    const float* b_ih       = (const float*)d_in[9];
    const float* W_hh       = (const float*)d_in[10];
    const float* b_hh       = (const float*)d_in[11];
    const float* W_out      = (const float*)d_in[12];
    const float* b_out      = (const float*)d_in[13];
    float* out = (float*)d_out;

    char* ws = (char*)d_ws;
    size_t off = 0;
    bf16_t* Winp = (bf16_t*)(ws + off); off += (size_t)Hsz * FINP * 2;
    off = (off + 255) & ~(size_t)255;
    bf16_t* Wcat = (bf16_t*)(ws + off); off += (size_t)G4H * KREC * 2;
    off = (off + 255) & ~(size_t)255;
    float*  bcat = (float*) (ws + off); off += (size_t)G4H * 4;
    off = (off + 255) & ~(size_t)255;
    bf16_t* emb  = (bf16_t*)(ws + off);

    convert_kernel<<<512, 256, 0, stream>>>(W_in, W_ih, W_hh, b_ih, b_hh,
                                            Winp, Wcat, bcat);
    embed_kernel<<<(Bsz * Ssz) / 16, 128, 0, stream>>>(x, id_indices, id_emb,
                                                       b_in, Winp, emb);
    recurrent_kernel<<<Bsz / 16, 256, 0, stream>>>(x, lengths, emb, Wcat, bcat,
                                                   W_dec, b_dec, W_out, b_out, out);
}
